// NoEdgeAttrGAT_88149908783509
// MI455X (gfx1250) — compile-verified, matrix operand order checked by reading, not run on hardware
//
#include <hip/hip_runtime.h>
#include <hip/hip_bf16.h>
#include <math.h>

typedef __attribute__((ext_vector_type(16))) _Float16 v16h;
typedef __attribute__((ext_vector_type(8)))  _Float16 v8h;
typedef __attribute__((ext_vector_type(8)))  float    v8f;

#define NEG_SLOPE 0.2f
#define HF 64   // hidden / feature width (all layers 64-wide)

// -------------------- helpers --------------------
__device__ __forceinline__ void atomicMaxF(float* addr, float val) {
  // IEEE-754 monotone trick; valid with init = -inf
  if (val >= 0.0f) atomicMax((int*)addr, __float_as_int(val));
  else             atomicMin((unsigned int*)addr, __float_as_uint(val));
}

// -------------------- weight prep: f32 W[k][n] -> f16 Wt[n][k], 4 matrices --------------------
__global__ __launch_bounds__(256)
void prep_weights(const float* __restrict__ W0, const float* __restrict__ W1,
                  const float* __restrict__ W2, const float* __restrict__ LW,
                  _Float16* __restrict__ Wt) {
  int tid = blockIdx.x * blockDim.x + threadIdx.x;
  if (tid >= 4 * HF * HF) return;
  int m   = tid >> 12;       // which matrix
  int idx = tid & 4095;
  int n   = idx >> 6;        // output row  (column of W)
  int k   = idx & 63;        // output col  (row of W)
  const float* W = (m == 0) ? W0 : (m == 1) ? W1 : (m == 2) ? W2 : LW;
  Wt[tid] = (_Float16)W[k * HF + n];
}

// -------------------- WMMA GEMM: Hout[nrows x 64] = X[nrows x 64] * Wt^T (+bias) ----------
// one wave per 16-row strip, 4x (16x16) f32 accumulators, K=64 as 2 x 32 (f16 inputs)
__global__ __launch_bounds__(256)
void gemm64_wmma(const float* __restrict__ X, const _Float16* __restrict__ Wt,
                 const float* __restrict__ bias, float* __restrict__ Hout, int nrows) {
  const int gw   = (int)((blockIdx.x * (unsigned)blockDim.x + threadIdx.x) >> 5);
  const int lane = threadIdx.x & 31;
  const int row0 = gw << 4;
  if (row0 >= nrows) return;               // whole-wave guard (EXEC stays all-ones)
  const int half = lane >> 4;
  const int ln   = lane & 15;

  v8f acc[4] = {};

#pragma unroll
  for (int kk = 0; kk < 2; ++kk) {
    // A fragment: row M = ln, K-chunks [half*8 .. +7] and [16+half*8 .. +7] (f32 -> f16)
    const float* xp = X + (size_t)(row0 + ln) * HF + kk * 32 + half * 8;
    float4 a0 = *(const float4*)(xp);
    float4 a1 = *(const float4*)(xp + 4);
    float4 a2 = *(const float4*)(xp + 16);
    float4 a3 = *(const float4*)(xp + 20);
    v16h A;
    A[0]=(_Float16)a0.x;  A[1]=(_Float16)a0.y;  A[2]=(_Float16)a0.z;  A[3]=(_Float16)a0.w;
    A[4]=(_Float16)a1.x;  A[5]=(_Float16)a1.y;  A[6]=(_Float16)a1.z;  A[7]=(_Float16)a1.w;
    A[8]=(_Float16)a2.x;  A[9]=(_Float16)a2.y;  A[10]=(_Float16)a2.z; A[11]=(_Float16)a2.w;
    A[12]=(_Float16)a3.x; A[13]=(_Float16)a3.y; A[14]=(_Float16)a3.z; A[15]=(_Float16)a3.w;

#pragma unroll
    for (int t = 0; t < 4; ++t) {
      // B fragment from pre-transposed f16 weights: column N = t*16+ln, same K-chunking
      const _Float16* wp = Wt + (size_t)(t * 16 + ln) * HF + kk * 32 + half * 8;
      v8h blo = *(const v8h*)(wp);
      v8h bhi = *(const v8h*)(wp + 16);
      v16h B;
#pragma unroll
      for (int i = 0; i < 8; ++i) { B[i] = blo[i]; B[i + 8] = bhi[i]; }
      acc[t] = __builtin_amdgcn_wmma_f32_16x16x32_f16(false, A, false, B,
                                                      (short)0, acc[t], false, false);
    }
  }

  // C/D layout: lanes 0-15: VGPR r -> (M=r, N=ln); lanes 16-31: (M=8+r, N=ln)
#pragma unroll
  for (int t = 0; t < 4; ++t) {
#pragma unroll
    for (int r = 0; r < 8; ++r) {
      int row = row0 + r + half * 8;
      int col = t * 16 + ln;
      float v = acc[t][r];
      if (bias) v += bias[col];
      Hout[(size_t)row * HF + col] = v;
    }
  }
}

// -------------------- alpha_s/alpha_d: wave-per-node dual dot product --------------------
__global__ __launch_bounds__(256)
void alpha_kernel(const float* __restrict__ H, const float* __restrict__ a_src,
                  const float* __restrict__ a_dst, float* __restrict__ as,
                  float* __restrict__ ad, int n) {
  int wave = (int)((blockIdx.x * (unsigned)blockDim.x + threadIdx.x) >> 5);
  int lane = threadIdx.x & 31;
  if (wave >= n) return;
  float h0 = H[(size_t)wave * HF + lane];
  float h1 = H[(size_t)wave * HF + 32 + lane];
  float s = h0 * a_src[lane] + h1 * a_src[32 + lane];
  float d = h0 * a_dst[lane] + h1 * a_dst[32 + lane];
#pragma unroll
  for (int off = 16; off > 0; off >>= 1) {
    s += __shfl_down(s, off, 32);
    d += __shfl_down(d, off, 32);
  }
  if (lane == 0) { as[wave] = s; ad[wave] = d; }
}

// -------------------- per-layer init: out = bias broadcast, max=-inf, denom=0 ------------
__global__ __launch_bounds__(256)
void init_layer(float* __restrict__ out, const float* __restrict__ b,
                float* __restrict__ mmax, float* __restrict__ denom, int n) {
  int tid = blockIdx.x * blockDim.x + threadIdx.x;
  if (tid >= n * HF) return;
  out[tid] = b[tid & 63];
  if (tid < n) { mmax[tid] = -INFINITY; denom[tid] = 0.0f; }
}

// -------------------- edge pass 1: e = leakyrelu(as[s]+ad[d]); segment max over dst -----
__global__ __launch_bounds__(256)
void edge_max(const int* __restrict__ eidx, const float* __restrict__ as,
              const float* __restrict__ ad, float* __restrict__ ee,
              float* __restrict__ mmax, int E, int n) {
  int e = blockIdx.x * blockDim.x + threadIdx.x;
  if (e >= E + n) return;
  int s, d;
  if (e < E) { s = eidx[e]; d = eidx[E + e]; } else { s = d = e - E; }
  float v = as[s] + ad[d];
  v = (v > 0.0f) ? v : NEG_SLOPE * v;
  ee[e] = v;
  atomicMaxF(&mmax[d], v);
}

// -------------------- edge pass 2: ee = exp(e - m[dst]); segment sum --------------------
__global__ __launch_bounds__(256)
void edge_exp(const int* __restrict__ eidx, float* __restrict__ ee,
              const float* __restrict__ mmax, float* __restrict__ denom, int E, int n) {
  int e = blockIdx.x * blockDim.x + threadIdx.x;
  if (e >= E + n) return;
  int d = (e < E) ? eidx[E + e] : (e - E);
  float v = __expf(ee[e] - mmax[d]);
  ee[e] = v;
  atomicAdd(&denom[d], v);
}

// -------------------- edge pass 3: out[dst] += (ee/denom[dst]) * h[src] -----------------
// 16 threads per edge, 4 features each (float4 gather + 4 f32 atomics)
__global__ __launch_bounds__(256)
void edge_scatter(const int* __restrict__ eidx, const float* __restrict__ ee,
                  const float* __restrict__ denom, const float* __restrict__ H,
                  float* __restrict__ out, int E, int n) {
  int tid = blockIdx.x * blockDim.x + threadIdx.x;
  if (tid >= (E + n) * 16) return;
  int e = tid >> 4;
  int f = (tid & 15) << 2;
  int s, d;
  if (e < E) { s = eidx[e]; d = eidx[E + e]; } else { s = d = e - E; }
  float coef = ee[e] / denom[d];
  float4 hv = *(const float4*)(H + (size_t)s * HF + f);
  float* op = out + (size_t)d * HF + f;
  atomicAdd(op + 0, coef * hv.x);
  atomicAdd(op + 1, coef * hv.y);
  atomicAdd(op + 2, coef * hv.z);
  atomicAdd(op + 3, coef * hv.w);
}

__global__ __launch_bounds__(256)
void relu_kernel(float* __restrict__ x, int n) {
  int tid = blockIdx.x * blockDim.x + threadIdx.x;
  if (tid < n) x[tid] = fmaxf(x[tid], 0.0f);
}

// -------------------- pooling --------------------
__global__ __launch_bounds__(256)
void init_pool(float* __restrict__ gmax, float* __restrict__ dg,
               float* __restrict__ pooled, int G) {
  int tid = blockIdx.x * blockDim.x + threadIdx.x;
  if (tid < G) { gmax[tid] = -INFINITY; dg[tid] = 0.0f; }
  if (tid < G * HF) pooled[tid] = 0.0f;
}

// gate score per node (wave-per-node dot) + per-graph max
__global__ __launch_bounds__(256)
void gate_kernel(const float* __restrict__ H, const float* __restrict__ gw,
                 const float* __restrict__ gb, const int* __restrict__ batch,
                 float* __restrict__ g, float* __restrict__ gmax, int n) {
  int wave = (int)((blockIdx.x * (unsigned)blockDim.x + threadIdx.x) >> 5);
  int lane = threadIdx.x & 31;
  if (wave >= n) return;
  float h0 = H[(size_t)wave * HF + lane];
  float h1 = H[(size_t)wave * HF + 32 + lane];
  float s = h0 * gw[lane] + h1 * gw[32 + lane];
#pragma unroll
  for (int off = 16; off > 0; off >>= 1) s += __shfl_down(s, off, 32);
  if (lane == 0) {
    s += gb[0];
    g[wave] = s;
    atomicMaxF(&gmax[batch[wave]], s);
  }
}

__global__ __launch_bounds__(256)
void gate_exp(const float* __restrict__ g, const float* __restrict__ gmax,
              const int* __restrict__ batch, float* __restrict__ eg,
              float* __restrict__ dg, int n) {
  int i = blockIdx.x * blockDim.x + threadIdx.x;
  if (i >= n) return;
  int b = batch[i];
  float v = __expf(g[i] - gmax[b]);
  eg[i] = v;
  atomicAdd(&dg[b], v);
}

// graph start offsets from sorted batch (handles empty graphs)
__global__ __launch_bounds__(256)
void bounds_kernel(const int* __restrict__ batch, int* __restrict__ gstart, int n, int G) {
  int i = blockIdx.x * blockDim.x + threadIdx.x;
  if (i >= n) return;
  int bc = batch[i];
  int bp = (i == 0) ? -1 : batch[i - 1];
  for (int q = bp + 1; q <= bc; ++q) gstart[q] = i;
  if (i == n - 1) for (int q = bc + 1; q <= G; ++q) gstart[q] = n;
}

// block-per-graph, thread-per-feature weighted sum (no atomics, fully coalesced)
__global__ __launch_bounds__(64)
void pool_kernel(const float* __restrict__ H, const float* __restrict__ eg,
                 const float* __restrict__ dg, const int* __restrict__ gstart,
                 float* __restrict__ pooled) {
  int g = blockIdx.x;
  int f = threadIdx.x;
  int s = gstart[g], e = gstart[g + 1];
  float dv  = dg[g];
  float inv = (dv > 0.0f) ? (1.0f / dv) : 0.0f;
  float acc = 0.0f;
  for (int nn = s; nn < e; ++nn) acc += eg[nn] * H[(size_t)nn * HF + f];
  pooled[g * HF + f] = acc * inv;
}

// -------------------- launch --------------------
extern "C" void kernel_launch(void* const* d_in, const int* in_sizes, int n_in,
                              void* d_out, int out_size, void* d_ws, size_t ws_size,
                              hipStream_t stream) {
  const float* x     = (const float*)d_in[0];
  const int*   eidx  = (const int*)d_in[1];
  const int*   batch = (const int*)d_in[2];
  const float* W[3]    = {(const float*)d_in[3], (const float*)d_in[7],  (const float*)d_in[11]};
  const float* asrc[3] = {(const float*)d_in[4], (const float*)d_in[8],  (const float*)d_in[12]};
  const float* adst[3] = {(const float*)d_in[5], (const float*)d_in[9],  (const float*)d_in[13]};
  const float* bl[3]   = {(const float*)d_in[6], (const float*)d_in[10], (const float*)d_in[14]};
  const float* gate_w = (const float*)d_in[15];
  const float* gate_b = (const float*)d_in[16];
  const float* lin_w  = (const float*)d_in[17];
  const float* lin_b  = (const float*)d_in[18];
  float* out = (float*)d_out;

  const int N  = in_sizes[2];
  const int E  = in_sizes[1] / 2;
  const int G  = 128;
  const int EL = E + N;

  // carve workspace (256B-aligned slices)
  char* p = (char*)d_ws;
  auto carve = [&](size_t bytes) -> char* {
    char* r = p; p += (bytes + 255) & ~(size_t)255; return r;
  };
  float*    hbuf0 = (float*)carve((size_t)N * HF * 4);  // GEMM output
  float*    hbuf1 = (float*)carve((size_t)N * HF * 4);  // aggregation output / layer input
  float*    as    = (float*)carve((size_t)N * 4);
  float*    ad    = (float*)carve((size_t)N * 4);
  float*    mmax  = (float*)carve((size_t)N * 4);
  float*    denom = (float*)carve((size_t)N * 4);
  float*    ee    = (float*)carve((size_t)EL * 4);
  float*    gsc   = (float*)carve((size_t)N * 4);
  float*    eg    = (float*)carve((size_t)N * 4);
  float*    gmax  = (float*)carve((size_t)G * 4);
  float*    dg    = (float*)carve((size_t)G * 4);
  int*      gst   = (int*)carve((size_t)(G + 1) * 4);
  float*    pool  = (float*)carve((size_t)G * HF * 4);
  _Float16* Wt    = (_Float16*)carve((size_t)4 * HF * HF * 2);

  const int B256 = 256;
  auto blk = [](long long n, int b) { return (int)((n + b - 1) / b); };

  prep_weights<<<blk(4 * HF * HF, B256), B256, 0, stream>>>(W[0], W[1], W[2], lin_w, Wt);

  for (int l = 0; l < 3; ++l) {
    const float* Xin = (l == 0) ? x : hbuf1;
    int waves  = N / 16;                              // N is a multiple of 16
    int gblk   = blk((long long)waves * 32, B256);
    gemm64_wmma<<<gblk, B256, 0, stream>>>(Xin, Wt + (size_t)l * HF * HF, nullptr, hbuf0, N);
    alpha_kernel<<<blk((long long)N * 32, B256), B256, 0, stream>>>(hbuf0, asrc[l], adst[l], as, ad, N);
    init_layer<<<blk((long long)N * HF, B256), B256, 0, stream>>>(hbuf1, bl[l], mmax, denom, N);
    edge_max<<<blk(EL, B256), B256, 0, stream>>>(eidx, as, ad, ee, mmax, E, N);
    edge_exp<<<blk(EL, B256), B256, 0, stream>>>(eidx, ee, mmax, denom, E, N);
    edge_scatter<<<blk((long long)EL * 16, B256), B256, 0, stream>>>(eidx, ee, denom, hbuf0, hbuf1, E, N);
    if (l < 2) relu_kernel<<<blk((long long)N * HF, B256), B256, 0, stream>>>(hbuf1, N * HF);
  }

  // global attention pooling over sorted batch
  init_pool<<<blk(G * HF, B256), B256, 0, stream>>>(gmax, dg, pool, G);
  gate_kernel<<<blk((long long)N * 32, B256), B256, 0, stream>>>(hbuf1, gate_w, gate_b, batch, gsc, gmax, N);
  gate_exp<<<blk(N, B256), B256, 0, stream>>>(gsc, gmax, batch, eg, dg, N);
  bounds_kernel<<<blk(N, B256), B256, 0, stream>>>(batch, gst, N, G);
  pool_kernel<<<G, HF, 0, stream>>>(hbuf1, eg, dg, gst, pool);

  // final linear: out[G x 64] = pooled @ lin_w + lin_b  (WMMA again, G=128 -> 8 waves)
  gemm64_wmma<<<1, B256, 0, stream>>>(pool, Wt + (size_t)3 * HF * HF, lin_b, out, G);
}
